// Decode_36197984371095
// MI455X (gfx1250) — matrix operand order repaired and checked by reading, not compiled, not run on hardware
//
#include <hip/hip_runtime.h>

typedef _Float16 f16;
typedef __attribute__((ext_vector_type(16))) _Float16 v16h;
typedef __attribute__((ext_vector_type(8)))  float    v8f;

#define HH 128
#define WW 128
#define BB 4
#define CIN 64
#define C1 256
#define NCTR 1024
#define NPTS 129            // P+1
#define K1 576              // 64*9
#define KT1 18
#define NT1 16              // 256/16
#define KT2 8               // K=256
#define NT2 4               // 64/16
#define KTG1 258            // K=8256
#define NTG1 32             // N=512
#define KTG2 16             // K=512
#define NTG2 16             // N=256

// ---- CDNA5 WMMA fragment swizzles (ISA 7.12.2, 16-bit operands, wave32) ----
// A (16xK tile, per lane 16 halfs): m = lane%16; K = (j/8)*16 + (lane/16)*8 + j%8
__device__ __forceinline__ int a_swz(int m16, int kk) {
    int lane = m16 + (((kk & 15) >> 3) << 4);
    int j    = ((kk >> 4) << 3) + (kk & 7);
    return lane * 16 + j;
}
// B (Kx16 tile): n = lane%16; K = (j/8)*16 + (lane/16)*8 + j%8
__device__ __forceinline__ int b_swz(int n16, int kk) {
    int lane = n16 + (((kk & 15) >> 3) << 4);
    int j    = ((kk >> 4) << 3) + (kk & 7);
    return lane * 16 + j;
}

__device__ __forceinline__ v8f wmma16(v16h a, v16h b, v8f c) {
    return __builtin_amdgcn_wmma_f32_16x16x32_f16(false, a, false, b, (short)0, c, false, false);
}

// ---------------- weight repack kernels ----------------
// Row-major W[N][K] (i.e. out = act @ W.T) -> B-tiles [kt][nt][lane][j] f16
__global__ void pack_b_rowmajor(const float* __restrict__ W, f16* __restrict__ dst,
                                int N, int K) {
    int e = blockIdx.x * blockDim.x + threadIdx.x;
    if (e >= N * K) return;
    int n = e / K, k = e % K;
    int kt = k >> 5, kk = k & 31, nt = n >> 4, NT = N >> 4;
    dst[(kt * NT + nt) * 512 + b_swz(n & 15, kk)] = (f16)W[e];
}

// conv1_w[o][c][ky][kx] -> B[k = (ky*3+kx)*64 + c][n = o], K=576, N=256
__global__ void pack_conv1(const float* __restrict__ W, f16* __restrict__ dst) {
    int e = blockIdx.x * blockDim.x + threadIdx.x;
    if (e >= C1 * K1) return;
    int o = e / K1, rem = e % K1;
    int c = rem / 9, t = rem % 9;
    int k = t * 64 + c;
    int kt = k >> 5, kk = k & 31, nt = o >> 4;
    dst[(kt * NT1 + nt) * 512 + b_swz(o & 15, kk)] = (f16)W[e];
}

// ---------------- fused conv1(3x3)+ReLU+conv2(1x1), f out in NHWC f32 -------
__global__ __launch_bounds__(256) void conv_fused(
    const float* __restrict__ x, const f16* __restrict__ WB1,
    const float* __restrict__ b1, const f16* __restrict__ WB2,
    const float* __restrict__ b2, float* __restrict__ fout) {
    __shared__ __align__(32) f16 sA [2 * KT1 * 512];   // 36 KB: im2col, A-swizzled
    __shared__ __align__(32) f16 sA2[2 * KT2 * 512];   // 16 KB: stage-2 activations
    int bid = blockIdx.x;
    int q = bid & 3, y = (bid >> 2) & 127, b = bid >> 9;
    int x0 = q * 32;
    int tid = threadIdx.x;

    // build im2col tile (32 px x 576) directly swizzled; coalesced reads over x
    for (int idx = tid; idx < 32 * K1; idx += 256) {
        int m = idx & 31;
        int c = (idx >> 5) & 63;
        int t = idx >> 11;                 // 0..8
        int dy = t / 3 - 1, dx = t % 3 - 1;
        int yy = y + dy, xx = x0 + m + dx;
        float v = 0.f;
        if (yy >= 0 && yy < HH && xx >= 0 && xx < WW)
            v = x[((b * CIN + c) * HH + yy) * WW + xx];
        int k = t * 64 + c;
        sA[((m >> 4) * KT1 + (k >> 5)) * 512 + a_swz(m & 15, k & 31)] = (f16)v;
    }
    __syncthreads();

    int wave = tid >> 5, lane = tid & 31;
    int mhi = (lane & 16) ? 8 : 0;
    {   // stage 1: 2 mtiles x 16 ntiles over 8 waves (1 mt x 4 nt each)
        int mt = wave & 1, ng = wave >> 1;
        v8f acc[4] = {};
        for (int kt = 0; kt < KT1; ++kt) {
            v16h a = *(const v16h*)&sA[(mt * KT1 + kt) * 512 + lane * 16];
            for (int i = 0; i < 4; ++i) {
                v16h bm = *(const v16h*)&WB1[(kt * NT1 + ng * 4 + i) * 512 + lane * 16];
                acc[i] = wmma16(a, bm, acc[i]);
            }
        }
        for (int i = 0; i < 4; ++i) {
            int ch = (ng * 4 + i) * 16 + (lane & 15);
            float bias = b1[ch];
            int kt2 = ch >> 5, kk2 = ch & 31;
            for (int r = 0; r < 8; ++r) {
                int m = mt * 16 + r + mhi;
                float v = acc[i][r] + bias;
                v = v > 0.f ? v : 0.f;
                sA2[((m >> 4) * KT2 + kt2) * 512 + a_swz(m & 15, kk2)] = (f16)v;
            }
        }
    }
    __syncthreads();
    {   // stage 2: 2 mtiles x 4 ntiles = 8 tiles, one per wave
        int mt = wave & 1, nt = wave >> 1;
        v8f acc = {};
        for (int kt = 0; kt < KT2; ++kt) {
            v16h a  = *(const v16h*)&sA2[(mt * KT2 + kt) * 512 + lane * 16];
            v16h bm = *(const v16h*)&WB2[(kt * NT2 + nt) * 512 + lane * 16];
            acc = wmma16(a, bm, acc);
        }
        int ch = nt * 16 + (lane & 15);
        float bias = b2[ch];
        for (int r = 0; r < 8; ++r) {
            int m = mt * 16 + r + mhi;
            fout[((b * HH + y) * WW + (x0 + m)) * 64 + ch] = acc[r] + bias;
        }
    }
}

// ------ init polys + bilinear sample -> fp (1024 x 8256) swizzled-A f16 -----
__global__ __launch_bounds__(128) void sample_build(
    const float* __restrict__ wh, const float* __restrict__ fmap,
    const int* __restrict__ ct_ind, const int* __restrict__ ct_img,
    f16* __restrict__ fpA, float* __restrict__ out_init) {
    __shared__ float spx[NPTS], spy[NPTS];
    int n = blockIdx.x, tid = threadIdx.x;
    int ci = ct_ind[n], b = ct_img[n];
    int ctx = ci % WW, cty = ci / WW;
    if (tid == 0) { spx[0] = (float)ctx; spy[0] = (float)cty; }
    {   // p = tid : init poly + output #1 (init_polys * DOWN)
        int p = tid;
        float ox = wh[((b * 256 + 2 * p    ) * HH + cty) * WW + ctx];
        float oy = wh[((b * 256 + 2 * p + 1) * HH + cty) * WW + ctx];
        float ipx = ox * 10.0f + (float)ctx;
        float ipy = oy * 10.0f + (float)cty;
        spx[1 + p] = ipx; spy[1 + p] = ipy;
        out_init[n * 256 + 2 * p]     = ipx * 4.0f;
        out_init[n * 256 + 2 * p + 1] = ipy * 4.0f;
    }
    __syncthreads();
    int c = tid & 63, ph = tid >> 6;
    for (int it = 0; it < 65; ++it) {
        int p = it * 2 + ph;
        if (p < NPTS) {
            // grid_sample(align_corners=False) collapses to (px-0.5, py-0.5)
            float sx = spx[p] - 0.5f, sy = spy[p] - 0.5f;
            float fx = floorf(sx),    fy = floorf(sy);
            int   xi = (int)fx,       yi = (int)fy;
            float wx = sx - fx,       wy = sy - fy;
            float v = 0.f;
            for (int q = 0; q < 4; ++q) {
                int yq = yi + (q >> 1), xq = xi + (q & 1);
                if (yq >= 0 && yq < HH && xq >= 0 && xq < WW) {
                    float w = ((q & 1) ? wx : 1.f - wx) * ((q >> 1) ? wy : 1.f - wy);
                    v += w * fmap[((b * HH + yq) * WW + xq) * 64 + c];
                }
            }
            int k = c * NPTS + p;                     // fp column: ch-major
            fpA[((n >> 4) * KTG1 + (k >> 5)) * 512 + a_swz(n & 15, k & 31)] = (f16)v;
        }
    }
}

// ------------- GEMM1: (1024x8256)@(8256x512), out swizzled-A f16 ------------
__global__ __launch_bounds__(256) void gemm1(
    const f16* __restrict__ fpA, const f16* __restrict__ WBP,
    f16* __restrict__ o1A) {
    int tid = threadIdx.x, wave = tid >> 5, lane = tid & 31;
    int mt  = blockIdx.x * 4 + (wave & 3);
    int ntb = blockIdx.y * 4 + ((wave >> 2) << 1);
    v8f acc[2] = {};
    for (int kt = 0; kt < KTG1; ++kt) {
        if (kt + 1 < KTG1)
            __builtin_prefetch(&WBP[((kt + 1) * NTG1 + ntb) * 512 + lane * 16], 0, 3);
        v16h a = *(const v16h*)&fpA[(mt * KTG1 + kt) * 512 + lane * 16];
        for (int i = 0; i < 2; ++i) {
            v16h bm = *(const v16h*)&WBP[(kt * NTG1 + ntb + i) * 512 + lane * 16];
            acc[i] = wmma16(a, bm, acc[i]);
        }
    }
    int mhi = (lane & 16) ? 8 : 0;
    for (int i = 0; i < 2; ++i) {
        int ncol = (ntb + i) * 16 + (lane & 15);     // becomes K of GEMM2
        int kt2 = ncol >> 5, kk2 = ncol & 31;
        for (int r = 0; r < 8; ++r) {
            int m = mt * 16 + r + mhi;
            o1A[((m >> 4) * KTG2 + kt2) * 512 + a_swz(m & 15, kk2)] = (f16)acc[i][r];
        }
    }
}

// -- GEMM2: (1024x512)@(512x256) + fuse_b, fuse *16 + init_polys*4 -> output -
__global__ __launch_bounds__(256) void gemm2(
    const f16* __restrict__ o1A, const f16* __restrict__ WBF,
    const float* __restrict__ fb, const float* __restrict__ init4,
    float* __restrict__ outc) {
    int tid = threadIdx.x, wave = tid >> 5, lane = tid & 31;
    int mt  = blockIdx.x * 4 + (wave & 3);
    int ntb = blockIdx.y * 4 + ((wave >> 2) << 1);
    v8f acc[2] = {};
    for (int kt = 0; kt < KTG2; ++kt) {
        v16h a = *(const v16h*)&o1A[(mt * KTG2 + kt) * 512 + lane * 16];
        for (int i = 0; i < 2; ++i) {
            v16h bm = *(const v16h*)&WBF[(kt * NTG2 + ntb + i) * 512 + lane * 16];
            acc[i] = wmma16(a, bm, acc[i]);
        }
    }
    int mhi = (lane & 16) ? 8 : 0;
    for (int i = 0; i < 2; ++i) {
        int col = (ntb + i) * 16 + (lane & 15);
        float bias = fb[col];
        for (int r = 0; r < 8; ++r) {
            int m = mt * 16 + r + mhi;
            // coarse*DOWN = (gemm+bias)*COARSE_STRIDE*DOWN + init_polys*DOWN
            outc[m * 256 + col] = (acc[i][r] + bias) * 16.0f + init4[m * 256 + col];
        }
    }
}

// ---------------------------------------------------------------------------
extern "C" void kernel_launch(void* const* d_in, const int* in_sizes, int n_in,
                              void* d_out, int out_size, void* d_ws, size_t ws_size,
                              hipStream_t stream) {
    const float* cnn    = (const float*)d_in[0];
    const float* wh     = (const float*)d_in[1];
    const float* c1w    = (const float*)d_in[2];
    const float* c1b    = (const float*)d_in[3];
    const float* c2w    = (const float*)d_in[4];
    const float* c2b    = (const float*)d_in[5];
    const float* polyw  = (const float*)d_in[6];
    const float* fusew  = (const float*)d_in[7];
    const float* fuseb  = (const float*)d_in[8];
    const int*   ct_ind = (const int*)d_in[9];
    const int*   ct_img = (const int*)d_in[10];
    float* out = (float*)d_out;

    char* ws = (char*)d_ws;
    f16*   WB1 = (f16*)(ws);                       // 576*256*2   = 294912
    f16*   WB2 = (f16*)(ws + 294912);              // 256*64*2    = 32768
    f16*   WBP = (f16*)(ws + 327680);              // 8256*512*2  = 8454144
    f16*   WBF = (f16*)(ws + 8781824);             // 512*256*2   = 262144
    float* F   = (float*)(ws + 9043968);           // 4*128*128*64*4 = 16777216 (NHWC)
    f16*   FPA = (f16*)(ws + 25821184);            // 1024*8256*2 = 16908288
    f16*   O1A = (f16*)(ws + 42729472);            // 1024*512*2  = 1048576
    (void)in_sizes; (void)n_in; (void)out_size; (void)ws_size;

    // weight repacks
    pack_conv1    <<<(C1 * K1 + 255) / 256, 256, 0, stream>>>(c1w, WB1);
    pack_b_rowmajor<<<(64 * 256 + 255) / 256, 256, 0, stream>>>(c2w, WB2, 64, 256);
    pack_b_rowmajor<<<(512 * 8256 + 255) / 256, 256, 0, stream>>>(polyw, WBP, 512, 8256);
    pack_b_rowmajor<<<(256 * 512 + 255) / 256, 256, 0, stream>>>(fusew, WBF, 256, 512);

    // fused conv1+relu+conv2 -> F (NHWC)
    conv_fused<<<BB * HH * (WW / 32), 256, 0, stream>>>(cnn, WB1, c1b, WB2, c2b, F);

    // init polys + bilinear gather -> FPA; writes d_out[0 : 1024*256]
    sample_build<<<NCTR, 128, 0, stream>>>(wh, F, ct_ind, ct_img, FPA, out);

    // fp @ poly_w.T  (1024x512)
    gemm1<<<dim3(16, 8), 256, 0, stream>>>(FPA, WBP, O1A);

    // @ fuse_w.T + b, fuse epilogue -> d_out[1024*256 : 2*1024*256]
    gemm2<<<dim3(16, 4), 256, 0, stream>>>(O1A, WBF, fuseb, out, out + 262144);
}
